// DotProductAttention_34385508172250
// MI455X (gfx1250) — matrix operand order repaired and checked by reading, not compiled, not run on hardware
//
#include <hip/hip_runtime.h>
#include <cstdint>
#include <cstddef>

namespace {

constexpr int kB  = 16;
constexpr int kLQ = 2048;
constexpr int kLV = 2048;
constexpr int kD  = 1024;

constexpr int kWaves   = 8;
constexpr int kThreads = kWaves * 32;     // 8 wave32 waves
constexpr int kKV      = 32;              // KV rows per flash iteration
constexpr int kKVB     = kLV / kKV;       // 64 KV blocks per batch
constexpr int kDSlice  = kD / kWaves;     // 128 d-columns per wave

// ---- workspace layout (filled by prep kernel) ----
constexpr size_t kWsVhiN = 0;                                   // [b][v][d] bf16 hi (64 MB)
constexpr size_t kWsVloN = kWsVhiN + (size_t)kB * kLV * kD * 2; // [b][v][d] bf16 lo (64 MB)
constexpr size_t kWsVhiT = kWsVloN + (size_t)kB * kLV * kD * 2; // [b][kvb][d][v32] hi (64 MB)
constexpr size_t kWsNeed = kWsVhiT + (size_t)kB * kLV * kD * 2; // 192 MB

constexpr size_t kPrepSmem = (size_t)kKV * kD * 2;              // 64 KB transpose staging

// LDS bytes: [V staging (fallback only)][Qh][Ql][Sp][pmax][psum][Pb][stats]
constexpr size_t smem_bytes(bool pre) {
  const size_t qr = pre ? 32 : 16;                 // q rows per block
  size_t s = pre ? 0 : (size_t)3 * 64 * 1024;      // VnH/VnL/VtH staging
  s += 2 * qr * (size_t)kD * 2;                    // Qh + Ql
  s += (size_t)kWaves * qr * kKV * 4;              // per-wave score partials
  s += 2 * qr * 8 * 4;                             // pmax + psum
  s += qr * (size_t)kKV * 2;                       // probs bf16
  s += 3 * qr * 4;                                 // m / l / alpha
  return s;
}
static_assert(smem_bytes(true)  <= 320u * 1024u, "kPre LDS over budget");
static_assert(smem_bytes(false) <= 320u * 1024u, "fallback LDS over budget");

typedef __attribute__((ext_vector_type(16))) __bf16 v16bf;
typedef __attribute__((ext_vector_type(8)))  float  v8f;

union BFrag {
  v16bf    v;
  uint16_t u[16];
  uint4    q[2];
};

__device__ __forceinline__ uint32_t f32_bits(float f) {
  union { float f; uint32_t u; } x; x.f = f; return x.u;
}
__device__ __forceinline__ float bits_f32(uint32_t u) {
  union { uint32_t u; float f; } x; x.u = u; return x.f;
}
__device__ __forceinline__ uint16_t bf16_trunc(float f) {
  return (uint16_t)(f32_bits(f) >> 16);
}

// Truncation split of 4 floats into packed bf16 hi / lo pairs.
// f = hi + lo, |err| ~ 2^-16 relative (bf16^2 class, plenty for exp()).
__device__ __forceinline__ void split_pack4(float4 f, uint2& ph, uint2& pl) {
  const uint32_t u0 = f32_bits(f.x), u1 = f32_bits(f.y);
  const uint32_t u2 = f32_bits(f.z), u3 = f32_bits(f.w);
  const uint16_t e0 = bf16_trunc(f.x - bits_f32(u0 & 0xFFFF0000u));
  const uint16_t e1 = bf16_trunc(f.y - bits_f32(u1 & 0xFFFF0000u));
  const uint16_t e2 = bf16_trunc(f.z - bits_f32(u2 & 0xFFFF0000u));
  const uint16_t e3 = bf16_trunc(f.w - bits_f32(u3 & 0xFFFF0000u));
  ph.x = (u0 >> 16) | (u1 & 0xFFFF0000u);
  ph.y = (u2 >> 16) | (u3 & 0xFFFF0000u);
  pl.x = (uint32_t)e0 | ((uint32_t)e1 << 16);
  pl.y = (uint32_t)e2 | ((uint32_t)e3 << 16);
}

// B fragment (32x16): lane holds 16 contiguous elements.
__device__ __forceinline__ BFrag load_b(const uint16_t* p) {
  BFrag r;
  r.q[0] = *(const uint4*)(p);
  r.q[1] = *(const uint4*)(p + 8);
  return r;
}
// A fragment (16x32): lane holds 16 contiguous K kbase..kbase+15.
__device__ __forceinline__ BFrag load_a(const uint16_t* p) {
  BFrag r;
  r.q[0] = *(const uint4*)(p);
  r.q[1] = *(const uint4*)(p + 8);
  return r;
}

__device__ __forceinline__ v8f wmma_bf16(v16bf a, v16bf b, v8f c) {
  return __builtin_amdgcn_wmma_f32_16x16x32_bf16(false, a, false, b, (short)0, c,
                                                 false, false);
}

}  // namespace

// ============================================================================
// Prep kernel: one workgroup per (batch, 32-row KV block). Converts V fp32 ->
// bf16 hi/lo row-major + hi transposed (per-block [d][v32]), coalesced stores.
// ============================================================================
__global__ void __launch_bounds__(kThreads, 1)
DotProductAttention_34385508172250_prep(const float* __restrict__ V,
                                        uint16_t* __restrict__ VhiN,
                                        uint16_t* __restrict__ VloN,
                                        uint16_t* __restrict__ VhiT) {
  extern __shared__ char psm[];
  uint16_t* Vt = (uint16_t*)psm;  // [d][v32]

  const int tid = threadIdx.x;
  const int b   = blockIdx.x / kKVB;
  const int kvb = blockIdx.x % kKVB;
  const int kv0 = kvb * kKV;

  const float* vsrc = V + ((size_t)b * kLV + kv0) * kD + tid * 4;
  uint16_t* hN = VhiN + ((size_t)b * kLV + kv0) * kD + tid * 4;
  uint16_t* lN = VloN + ((size_t)b * kLV + kv0) * kD + tid * 4;

#pragma unroll 4
  for (int i = 0; i < kKV; ++i) {
    float4 f = *(const float4*)(vsrc + (size_t)i * kD);
    uint2 ph, pl;
    split_pack4(f, ph, pl);
    *(uint2*)(hN + (size_t)i * kD) = ph;
    *(uint2*)(lN + (size_t)i * kD) = pl;
    const int d = tid * 4;
    Vt[(d + 0) * kKV + i] = (uint16_t)(ph.x & 0xFFFF);
    Vt[(d + 1) * kKV + i] = (uint16_t)(ph.x >> 16);
    Vt[(d + 2) * kKV + i] = (uint16_t)(ph.y & 0xFFFF);
    Vt[(d + 3) * kKV + i] = (uint16_t)(ph.y >> 16);
  }
  __syncthreads();

  uint4*       gT = (uint4*)(VhiT + ((size_t)(b * kKVB + kvb)) * kD * kKV);
  const uint4* sT = (const uint4*)Vt;
#pragma unroll
  for (int j = tid; j < (kD * kKV) / 8; j += kThreads) gT[j] = sT[j];
}

// ============================================================================
// Main flash-attention kernel.
//   kPre = true : M=32 (two M tiles share every B fragment -> half the L2
//                 traffic); B operands stream from pre-converted global arrays.
//   kPre = false: M=16 self-contained fallback, stages+converts V in LDS.
// Q hi/lo staged in LDS; softmax reduction parallelized across all waves.
// ============================================================================
template <bool kPre>
__global__ void __launch_bounds__(kThreads, 1)
DotProductAttention_34385508172250_kernel(const float* __restrict__ Q,
                                          const float* __restrict__ V,
                                          const uint16_t* __restrict__ VhiN,
                                          const uint16_t* __restrict__ VloN,
                                          const uint16_t* __restrict__ VhiT,
                                          float* __restrict__ O) {
  constexpr int MT = kPre ? 2 : 1;   // 16-row M tiles per block
  constexpr int QR = 16 * MT;        // q rows per block
  constexpr int RT = QR * 8;         // threads in parallel softmax (8 per row)

  extern __shared__ char smem[];
  constexpr size_t offQh = kPre ? 0 : (size_t)3 * 64 * 1024;
  constexpr size_t szQ   = (size_t)QR * kD * 2;
  constexpr size_t offQl = offQh + szQ;
  constexpr size_t offSp = offQl + szQ;
  constexpr size_t szSp  = (size_t)kWaves * QR * kKV * 4;
  constexpr size_t offPm = offSp + szSp;
  constexpr size_t szPm  = (size_t)RT * 4;
  constexpr size_t offPs = offPm + szPm;
  constexpr size_t offPb = offPs + szPm;
  constexpr size_t szPb  = (size_t)QR * kKV * 2;
  constexpr size_t offSt = offPb + szPb;

  uint16_t* Qh   = (uint16_t*)(smem + offQh);
  uint16_t* Ql   = (uint16_t*)(smem + offQl);
  float*    Sp   = (float*)   (smem + offSp);
  float*    pmax = (float*)   (smem + offPm);
  float*    psum = (float*)   (smem + offPs);
  uint16_t* Pb   = (uint16_t*)(smem + offPb);
  float*    mrow = (float*)   (smem + offSt);
  float*    lrow = mrow + QR;
  float*    arow = mrow + 2 * QR;
  uint16_t* VnH = nullptr;  // fallback-only LDS V staging
  uint16_t* VnL = nullptr;
  uint16_t* VtH = nullptr;
  if constexpr (!kPre) {
    VnH = (uint16_t*)(smem + 0 * 64 * 1024);
    VnL = (uint16_t*)(smem + 1 * 64 * 1024);
    VtH = (uint16_t*)(smem + 2 * 64 * 1024);
  }

  const int tid   = threadIdx.x;
  const int wave  = tid >> 5;
  const int lane  = tid & 31;
  const int lhalf = lane >> 4;   // lane half selects K/M sub-block per ISA striping
  const int l16   = lane & 15;
  const int b     = blockIdx.x / (kLQ / QR);
  const int q0    = (blockIdx.x % (kLQ / QR)) * QR;
  const int dbase = wave * kDSlice;

  if (tid < QR) { mrow[tid] = -INFINITY; lrow[tid] = 0.0f; }

  // ---- Stage Q hi/lo into LDS (row-major [QR][kD]) ----
  {
    const float* qsrc = Q + ((size_t)(b * kLQ + q0)) * kD + tid * 4;
    uint16_t* qhp = Qh + tid * 4;
    uint16_t* qlp = Ql + tid * 4;
#pragma unroll 4
    for (int i = 0; i < QR; ++i) {
      float4 f = *(const float4*)(qsrc + (size_t)i * kD);
      uint2 ph, pl;
      split_pack4(f, ph, pl);
      *(uint2*)(qhp + (size_t)i * kD) = ph;
      *(uint2*)(qlp + (size_t)i * kD) = pl;
    }
  }
  __syncthreads();

  v8f acc[MT][8];
#pragma unroll
  for (int mt = 0; mt < MT; ++mt)
#pragma unroll
    for (int t = 0; t < 8; ++t) acc[mt][t] = (v8f)0.0f;

  for (int kv0 = 0; kv0 < kLV; kv0 += kKV) {
    // ---- Phase A (fallback only): stage + convert V tile into LDS ----
    if constexpr (!kPre) {
      __syncthreads();  // previous iteration done reading V LDS
      const float* vsrc = V + ((size_t)b * kLV + kv0) * kD + tid * 4;
      __builtin_prefetch(vsrc + (size_t)kKV * kD, 0, 1);
#pragma unroll 4
      for (int i = 0; i < kKV; ++i) {
        float4 f = *(const float4*)(vsrc + (size_t)i * kD);
        uint2 ph, pl;
        split_pack4(f, ph, pl);
        *(uint2*)(VnH + (size_t)i * kD + tid * 4) = ph;
        *(uint2*)(VnL + (size_t)i * kD + tid * 4) = pl;
        const int d = tid * 4;
        VtH[(d + 0) * kKV + i] = (uint16_t)(ph.x & 0xFFFF);
        VtH[(d + 1) * kKV + i] = (uint16_t)(ph.x >> 16);
        VtH[(d + 2) * kKV + i] = (uint16_t)(ph.y & 0xFFFF);
        VtH[(d + 3) * kKV + i] = (uint16_t)(ph.y >> 16);
      }
      __syncthreads();
    } else if (kv0 + kKV < kLV) {
      __builtin_prefetch(VhiN + ((size_t)b * kLV + kv0 + kKV) * kD + lane * 64, 0, 1);
    }

    // ---- Phase B: partial scores over this wave's 128-d slice ----
    // B layout (32x16): lane col n=l16, K half = lhalf*16, 16 contiguous elems.
    const uint16_t* rowH = kPre ? (VhiN + ((size_t)b * kLV + kv0) * kD) : VnH;
    const uint16_t* rowL = kPre ? (VloN + ((size_t)b * kLV + kv0) * kD) : VnL;
    v8f s[MT][2];
#pragma unroll
    for (int mt = 0; mt < MT; ++mt) { s[mt][0] = (v8f)0.0f; s[mt][1] = (v8f)0.0f; }

#pragma unroll
    for (int kk = 0; kk < 4; ++kk) {
      const int coff = dbase + kk * 32 + lhalf * 16;
      const int aoff = dbase + kk * 32 + lhalf * 16;
      BFrag ah[MT], al[MT];
#pragma unroll
      for (int mt = 0; mt < MT; ++mt) {
        ah[mt] = load_a(Qh + (size_t)(mt * 16 + l16) * kD + aoff);
        al[mt] = load_a(Ql + (size_t)(mt * 16 + l16) * kD + aoff);
      }
      const BFrag bh0 = load_b(rowH + (size_t)l16 * kD + coff);
      const BFrag bl0 = load_b(rowL + (size_t)l16 * kD + coff);
      const BFrag bh1 = load_b(rowH + (size_t)(16 + l16) * kD + coff);
      const BFrag bl1 = load_b(rowL + (size_t)(16 + l16) * kD + coff);
      // hi*hi + hi*lo + lo*hi : ~fp32-accurate scores out of bf16 WMMA.
      // B fragments are shared by all M tiles (amortizes the global loads).
#pragma unroll
      for (int mt = 0; mt < MT; ++mt) {
        s[mt][0] = wmma_bf16(ah[mt].v, bh0.v, s[mt][0]);
        s[mt][0] = wmma_bf16(ah[mt].v, bl0.v, s[mt][0]);
        s[mt][0] = wmma_bf16(al[mt].v, bh0.v, s[mt][0]);
        s[mt][1] = wmma_bf16(ah[mt].v, bh1.v, s[mt][1]);
        s[mt][1] = wmma_bf16(ah[mt].v, bl1.v, s[mt][1]);
        s[mt][1] = wmma_bf16(al[mt].v, bh1.v, s[mt][1]);
      }
    }
    {  // C/D layout: lane n=l16, VGPR r -> row m = r + lhalf*8
      float* sp = Sp + (size_t)wave * QR * kKV;
#pragma unroll
      for (int mt = 0; mt < MT; ++mt)
#pragma unroll
        for (int r = 0; r < 8; ++r) {
          const int m = mt * 16 + r + lhalf * 8;
          sp[m * kKV + l16]      = s[mt][0][r];
          sp[m * kKV + 16 + l16] = s[mt][1][r];
        }
    }
    __syncthreads();

    // ---- Softmax, parallelized: thread owns (row = tid>>3, 4 cols) ----
    const int srow = tid >> 3;
    const int scol = (tid & 7) * 4;
    float4 s4;  // cross-wave reduced scores, kept in registers across barriers
    s4.x = s4.y = s4.z = s4.w = 0.0f;
    if (tid < RT) {
#pragma unroll
      for (int w = 0; w < kWaves; ++w) {
        const float4 p =
            *(const float4*)(Sp + (size_t)w * QR * kKV + srow * kKV + scol);
        s4.x += p.x; s4.y += p.y; s4.z += p.z; s4.w += p.w;
      }
      pmax[tid] = fmaxf(fmaxf(s4.x, s4.y), fmaxf(s4.z, s4.w));
    }
    __syncthreads();

    if (tid < QR) {  // fold 8 partial maxes per row; compute alpha
      float mx = mrow[tid];
#pragma unroll
      for (int p = 0; p < 8; ++p) mx = fmaxf(mx, pmax[tid * 8 + p]);
      arow[tid] = __expf(mrow[tid] - mx);  // 0 on first tile (mrow = -inf)
      mrow[tid] = mx;
    }
    __syncthreads();

    if (tid < RT) {  // exponentiate register-resident sums, emit bf16 probs
      const float mx = mrow[srow];
      const float e0 = __expf(s4.x - mx);
      const float e1 = __expf(s4.y - mx);
      const float e2 = __expf(s4.z - mx);
      const float e3 = __expf(s4.w - mx);
      psum[tid] = (e0 + e1) + (e2 + e3);
      uint2 pk;
      pk.x = (uint32_t)bf16_trunc(e0) | ((uint32_t)bf16_trunc(e1) << 16);
      pk.y = (uint32_t)bf16_trunc(e2) | ((uint32_t)bf16_trunc(e3) << 16);
      *(uint2*)(Pb + (size_t)srow * kKV + scol) = pk;
    }
    __syncthreads();

    // Denominator update: runs in wave 0 concurrently with Phase E issue;
    // lrow is only consumed after the post-loop barrier.
    if (tid < QR) {
      float sum = 0.0f;
#pragma unroll
      for (int p = 0; p < 8; ++p) sum += psum[tid * 8 + p];
      lrow[tid] = lrow[tid] * arow[tid] + sum;
    }

    // ---- Phase E: rescale accumulators; ctx += P(QRx32) @ V(32 x 128-slice) ----
    {
      float alr[MT][8];
#pragma unroll
      for (int mt = 0; mt < MT; ++mt)
#pragma unroll
        for (int r = 0; r < 8; ++r) alr[mt][r] = arow[mt * 16 + r + lhalf * 8];
#pragma unroll
      for (int mt = 0; mt < MT; ++mt)
#pragma unroll
        for (int t = 0; t < 8; ++t)
#pragma unroll
          for (int r = 0; r < 8; ++r) acc[mt][t][r] *= alr[mt][r];

      BFrag pf[MT];  // A fragments of P
#pragma unroll
      for (int mt = 0; mt < MT; ++mt)
        pf[mt] = load_a(Pb + (size_t)(mt * 16 + l16) * kKV + lhalf * 16);

      // B[k=kv][n=dcol] = Vt[dcol][kv]: 16 contiguous kv per lane.
      const uint16_t* tbase =
          kPre ? (VhiT + ((size_t)(b * kKVB + (kv0 / kKV))) * kD * kKV) : VtH;
#pragma unroll
      for (int t = 0; t < 8; ++t) {
        const BFrag bf =
            load_b(tbase + (size_t)(dbase + t * 16 + l16) * kKV + lhalf * 16);
#pragma unroll
        for (int mt = 0; mt < MT; ++mt)
          acc[mt][t] = wmma_bf16(pf[mt].v, bf.v, acc[mt][t]);
      }
    }
  }
  __syncthreads();  // lrow final values visible to all waves

  // ---- Epilogue: divide by softmax denominator, store fp32 ----
  {
    float linv[MT][8];
#pragma unroll
    for (int mt = 0; mt < MT; ++mt)
#pragma unroll
      for (int r = 0; r < 8; ++r)
        linv[mt][r] = 1.0f / lrow[mt * 16 + r + lhalf * 8];
#pragma unroll
    for (int mt = 0; mt < MT; ++mt)
#pragma unroll
      for (int t = 0; t < 8; ++t) {
        const int col = dbase + t * 16 + l16;
#pragma unroll
        for (int r = 0; r < 8; ++r) {
          const int m = mt * 16 + r + lhalf * 8;
          O[((size_t)(b * kLQ + q0 + m)) * kD + col] = acc[mt][t][r] * linv[mt][r];
        }
      }
  }
}

extern "C" void kernel_launch(void* const* d_in, const int* in_sizes, int n_in,
                              void* d_out, int out_size, void* d_ws, size_t ws_size,
                              hipStream_t stream) {
  (void)in_sizes; (void)n_in; (void)out_size;
  const float* Q = (const float*)d_in[0];
  const float* V = (const float*)d_in[1];
  float* O = (float*)d_out;
  const dim3 block(kThreads);  // 256 threads = 8 wave32

  if (d_ws != nullptr && ws_size >= kWsNeed) {
    uint16_t* VhiN = (uint16_t*)((char*)d_ws + kWsVhiN);
    uint16_t* VloN = (uint16_t*)((char*)d_ws + kWsVloN);
    uint16_t* VhiT = (uint16_t*)((char*)d_ws + kWsVhiT);
    DotProductAttention_34385508172250_prep
        <<<dim3(kB * kKVB), block, kPrepSmem, stream>>>(V, VhiN, VloN, VhiT);
    DotProductAttention_34385508172250_kernel<true>
        <<<dim3(kB * (kLQ / 32)), block, smem_bytes(true), stream>>>(
            Q, V, VhiN, VloN, VhiT, O);
  } else {
    DotProductAttention_34385508172250_kernel<false>
        <<<dim3(kB * (kLQ / 16)), block, smem_bytes(false), stream>>>(
            Q, V, nullptr, nullptr, nullptr, O);
  }
}
